// SelfAttention_11536282157500
// MI455X (gfx1250) — matrix operand order repaired and checked by reading, not compiled, not run on hardware
//
#include <hip/hip_runtime.h>

// Problem constants (from reference)
#define B_  4
#define S_  2048
#define H_  1024
#define NH_ 16
#define HD_ 64
#define M_  (B_ * S_)   // 8192 rows

typedef __bf16 bf16;
typedef __attribute__((ext_vector_type(16))) __bf16 v16bf;
typedef __attribute__((ext_vector_type(8)))  float  v8f;
typedef __attribute__((ext_vector_type(4)))  unsigned v4u;
typedef __attribute__((ext_vector_type(8)))  int    v8i;
typedef __attribute__((ext_vector_type(4)))  int    v4i;

__device__ __forceinline__ bf16 f2bf(float f) {
  union { float f; unsigned u; } v; v.f = f;
  unsigned r = v.u + 0x7FFFu + ((v.u >> 16) & 1u);   // round-to-nearest-even
  unsigned short h = (unsigned short)(r >> 16);
  union { unsigned short s; bf16 b; } o; o.s = h;
  return o.b;
}

// Generic pointer to __shared__ has the LDS byte offset in its low 32 bits
__device__ __forceinline__ unsigned lds_off(const void* p) {
  return (unsigned)(unsigned long long)p;
}

// ---------------------------------------------------------------------------
// Tensor Data Mover: DMA a 2D tile (tile_w x tile_h elements of bf16, global
// row stride `stride` elements) into LDS, adding `amt_code` dwords of padding
// every `ivl_code` dwords so the LDS layout lands on our padded row stride.
// D# layout per CDNA5 ISA ch.8: group0 = {count|flags, lds_addr, gaddr_lo,
// gaddr_hi|type=2}; group1 = {mask/data_size/pad, dims/tiles/strides}.
// Issued by one wave (EXEC ignored); tracked with TENSORcnt.
// ---------------------------------------------------------------------------
__device__ __forceinline__ void tdm_load_2d(const void* g, unsigned ldsoff,
                                            int tile_w, int tile_h, int stride,
                                            int ivl_code, int amt_code) {
  unsigned long long ga = (unsigned long long)g;
  v4u g0;
  g0.x = 1u;                                                // count=1, user mode
  g0.y = ldsoff;                                            // lds_addr
  g0.z = (unsigned)ga;                                      // global_addr[31:0]
  g0.w = (unsigned)((ga >> 32) & 0x01FFFFFFull) | (2u << 30); // addr[56:32], type=2
  v8i g1;
  g1[0] = (1 << 16)                 // data_size = 1 -> 2 bytes
        | (1 << 20)                 // pad_enable
        | (ivl_code << 22)          // pad_interval
        | (amt_code << 25);         // pad_amount
  g1[1] = (tile_w & 0xFFFF) << 16;                          // tensor_dim0[15:0] = tile_w
  g1[2] = (((unsigned)tile_w >> 16) & 0xFFFF)               // tensor_dim0[31:16]
        | ((tile_h & 0xFFFF) << 16);                        // tensor_dim1[15:0] = tile_h
  g1[3] = (((unsigned)tile_h >> 16) & 0xFFFF)               // tensor_dim1[31:16]
        | ((tile_w & 0xFFFF) << 16);                        // tile_dim0 = tile_w
  g1[4] = (tile_h & 0xFFFF);                                // tile_dim1 = tile_h, tile_dim2=0
  g1[5] = stride;                                           // tensor_dim0_stride[31:0]
  g1[6] = 0;                                                // stride hi, dim1_stride lo
  g1[7] = 0;
  v4i gz = {0, 0, 0, 0};
#if __clang_major__ >= 23
  v8i gz8 = {0, 0, 0, 0, 0, 0, 0, 0};
  __builtin_amdgcn_tensor_load_to_lds(g0, g1, gz, gz, gz8, 0);
#else
  __builtin_amdgcn_tensor_load_to_lds(g0, g1, gz, gz, 0);
#endif
}

// ---------------------------------------------------------------------------
// f32 -> bf16 conversion (grid-stride)
// ---------------------------------------------------------------------------
__global__ void cvt_f32_bf16(const float* __restrict__ in, bf16* __restrict__ out, int n) {
  int i = blockIdx.x * blockDim.x + threadIdx.x;
  int stride = gridDim.x * blockDim.x;
  for (; i < n; i += stride) out[i] = f2bf(in[i]);
}

// ---------------------------------------------------------------------------
// GEMM: out[m,n] = sum_k A[m,k] * W[n,k] + bias[n]      (y = x @ W.T + b)
// A: bf16 [Mdim,Kdim] row-major, W: bf16 [Ndim,Kdim] row-major.
// MODE 0: out = bf16 flat; MODE 1: out = f32 flat.
// Block: 256 threads = 8 waves; tile 256(M) x 64(N); K-step 32.
// Each wave owns 32 rows x 64 cols = 8x v_wmma_f32_16x16x32_bf16 per K-step.
// TDM double-buffered LDS staging: DMA tile t+1 while computing tile t.
// ---------------------------------------------------------------------------
template <int MODE>
__global__ __launch_bounds__(256)
void gemm_bf16(const bf16* __restrict__ A, const bf16* __restrict__ W,
               const float* __restrict__ bias, void* __restrict__ outp,
               int Mdim, int Ndim, int Kdim) {
  constexpr int MT = 256, NT = 64, KT = 32;
  constexpr int LDA = 40;                 // 32 elems + 4-dword TDM pad (80B rows)
  __shared__ bf16 As[2][MT][LDA];
  __shared__ bf16 Ws[2][NT][LDA];

  const int m0   = blockIdx.x * MT;
  const int n0   = blockIdx.y * NT;
  const int t    = threadIdx.x;
  const int wave = t >> 5;                // 0..7
  const int lane = t & 31;
  const int hi   = lane >> 4;
  const int ln   = lane & 15;

  v8f acc[2][4];
#pragma unroll
  for (int rg = 0; rg < 2; rg++)
#pragma unroll
    for (int i = 0; i < 4; i++)
#pragma unroll
      for (int e = 0; e < 8; e++) acc[rg][i][e] = 0.f;

  // Preload K-tile 0 via TDM (pad: every 16 dwords insert 4 dwords -> stride 40)
  if (wave == 0) {
    tdm_load_2d(A + (size_t)m0 * Kdim, lds_off(&As[0][0][0]), KT, MT, Kdim, 3, 3);
    tdm_load_2d(W + (size_t)n0 * Kdim, lds_off(&Ws[0][0][0]), KT, NT, Kdim, 3, 3);
    __builtin_amdgcn_s_wait_tensorcnt(0);
  }
  __syncthreads();

  const int nsteps = Kdim / KT;
  for (int s = 0; s < nsteps; s++) {
    const int cur = s & 1;
    const int nxt = cur ^ 1;
    if (wave == 0 && s + 1 < nsteps) {
      const int k0 = (s + 1) * KT;
      tdm_load_2d(A + (size_t)m0 * Kdim + k0, lds_off(&As[nxt][0][0]), KT, MT, Kdim, 3, 3);
      tdm_load_2d(W + (size_t)n0 * Kdim + k0, lds_off(&Ws[nxt][0][0]), KT, NT, Kdim, 3, 3);
    }

    // A fragments (ISA 16-bit A 16x32 layout): lane = row; K pairs split by lane half
    v16bf af[2];
#pragma unroll
    for (int rg = 0; rg < 2; rg++) {
      const int arow = wave * 32 + rg * 16 + ln;
#pragma unroll
      for (int p = 0; p < 8; p++) {
        int kk = (p < 4) ? (hi * 8 + 2 * p) : (16 + hi * 8 + 2 * (p - 4));
        af[rg][2 * p]     = As[cur][arow][kk];
        af[rg][2 * p + 1] = As[cur][arow][kk + 1];
      }
    }
    // B fragments shared by both row groups: lane = column; same k order as A (8-k groups by lane half)
#pragma unroll
    for (int nt = 0; nt < 4; nt++) {
      v16bf bfr;
      const int brow = nt * 16 + ln;
      const int kst  = hi * 16;
#pragma unroll
      for (int j = 0; j < 16; j++) bfr[j] = Ws[cur][brow][kst / 2 + j + (j & 8)];
      acc[0][nt] = __builtin_amdgcn_wmma_f32_16x16x32_bf16(
          false, af[0], false, bfr, (short)0, acc[0][nt], false, false);
      acc[1][nt] = __builtin_amdgcn_wmma_f32_16x16x32_bf16(
          false, af[1], false, bfr, (short)0, acc[1][nt], false, false);
    }

    if (wave == 0 && s + 1 < nsteps) __builtin_amdgcn_s_wait_tensorcnt(0);
    __syncthreads();
  }

  // Epilogue: C layout -> element r is row (hi*8 + r), col = ln per 16-col subtile
#pragma unroll
  for (int rg = 0; rg < 2; rg++)
#pragma unroll
    for (int nt = 0; nt < 4; nt++) {
      const int n = n0 + nt * 16 + ln;
      const float bv = bias[n];
#pragma unroll
      for (int r = 0; r < 8; r++) {
        const int m = m0 + wave * 32 + rg * 16 + hi * 8 + r;
        const float v = acc[rg][nt][r] + bv;
        if (MODE == 0) ((bf16*)outp)[(size_t)m * Ndim + n] = f2bf(v);
        else           ((float*)outp)[(size_t)m * Ndim + n] = v;
      }
    }
}

// ---------------------------------------------------------------------------
// Flash attention: one block = (b, h, 64-row Q tile); 128 threads = 4 waves.
// Q/K/V/ctx are bf16 flat [B*S, H]; head h occupies columns [h*64, h*64+64).
// Q and K tiles arrive via TDM; V is staged transposed by the waves.
// ---------------------------------------------------------------------------
__global__ __launch_bounds__(128)
void flash_attn(const bf16* __restrict__ Q, const bf16* __restrict__ K,
                const bf16* __restrict__ V, bf16* __restrict__ ctx) {
  constexpr int QT = 64, KTILE = 64;
  constexpr int LD = 72;                  // 64 elems + 4-dword TDM pad (144B rows)
  __shared__ bf16 Qs[QT][LD];
  __shared__ bf16 Ks[KTILE][LD];
  __shared__ bf16 Vt[HD_][LD];            // V transposed: Vt[d][key]
  __shared__ bf16 Ps[4][16][LD];          // per-wave P tile (C-layout -> A-layout hop)

  const int bh = blockIdx.x;              // 0..B*NH-1
  const int b  = bh / NH_;
  const int h  = bh % NH_;
  const int q0 = blockIdx.y * QT;
  const int t = threadIdx.x, wave = t >> 5, lane = t & 31;
  const int hi = lane >> 4, ln = lane & 15;
  const int hcol = h * HD_;

  // Q tile (64x64, row stride H) via TDM; pad every 32 dwords by 4 -> stride 72
  if (wave == 0) {
    tdm_load_2d(Q + (size_t)(b * S_ + q0) * H_ + hcol, lds_off(&Qs[0][0]),
                HD_, QT, H_, 4, 3);
    __builtin_amdgcn_s_wait_tensorcnt(0);
  }
  __syncthreads();

  // Q fragments for this wave (2 K-steps of 32), kept in registers
  v16bf qa[2];
  {
    const int arow = wave * 16 + ln;
#pragma unroll
    for (int ks = 0; ks < 2; ks++)
#pragma unroll
      for (int p = 0; p < 8; p++) {
        int kk = ks * 32 + ((p < 4) ? (hi * 8 + 2 * p) : (16 + hi * 8 + 2 * (p - 4)));
        qa[ks][2 * p]     = Qs[arow][kk];
        qa[ks][2 * p + 1] = Qs[arow][kk + 1];
      }
  }

  float row_m[8], row_l[8];
#pragma unroll
  for (int r = 0; r < 8; r++) { row_m[r] = -3.0e38f; row_l[r] = 0.f; }
  v8f o[4];
#pragma unroll
  for (int i = 0; i < 4; i++)
#pragma unroll
    for (int e = 0; e < 8; e++) o[i][e] = 0.f;

  for (int kc = 0; kc < S_ / KTILE; kc++) {
    // K chunk via TDM (wave 0) while all waves stage V transposed
    if (wave == 0)
      tdm_load_2d(K + (size_t)(b * S_ + kc * KTILE) * H_ + hcol, lds_off(&Ks[0][0]),
                  HD_, KTILE, H_, 4, 3);
    {
      const int row = t >> 1;
      const int col = (t & 1) * 32;
      const int4* v4 = (const int4*)(V + (size_t)(b * S_ + kc * KTILE + row) * H_ + hcol + col);
      int4 vv[4];
#pragma unroll
      for (int i = 0; i < 4; i++) vv[i] = v4[i];
      const bf16* pv = (const bf16*)vv;
#pragma unroll
      for (int j = 0; j < 32; j++) Vt[col + j][row] = pv[j];
    }
    if (wave == 0) __builtin_amdgcn_s_wait_tensorcnt(0);
    __syncthreads();

    // S = Q * K^T  (A = Q frag; B lane n = key row, K-dim = d)
    v8f sc[4];
#pragma unroll
    for (int i = 0; i < 4; i++)
#pragma unroll
      for (int e = 0; e < 8; e++) sc[i][e] = 0.f;
#pragma unroll
    for (int nt = 0; nt < 4; nt++) {
#pragma unroll
      for (int ks = 0; ks < 2; ks++) {
        v16bf bfr;
        const int krow = nt * 16 + ln;
        const int kst  = ks * 32 + hi * 16;
#pragma unroll
        for (int j = 0; j < 16; j++) bfr[j] = Ks[krow][(kst & ~31) + ((kst & 16) >> 1) + j + (j & 8)];
        sc[nt] = __builtin_amdgcn_wmma_f32_16x16x32_bf16(
            false, qa[ks], false, bfr, (short)0, sc[nt], false, false);
      }
    }

    // Online softmax. Row m = hi*8 + r; N spans the 16 lanes of each half-wave.
#pragma unroll
    for (int r = 0; r < 8; r++) {
      float mx = -3.0e38f;
#pragma unroll
      for (int nt = 0; nt < 4; nt++) mx = fmaxf(mx, sc[nt][r] * 0.125f);
      for (int off = 8; off; off >>= 1) mx = fmaxf(mx, __shfl_xor(mx, off, 32));
      const float nm   = fmaxf(row_m[r], mx);
      const float corr = __expf(row_m[r] - nm);
      float ps = 0.f;
#pragma unroll
      for (int nt = 0; nt < 4; nt++) {
        float p = __expf(sc[nt][r] * 0.125f - nm);
        sc[nt][r] = p;                      // reuse as P
        ps += p;
      }
      for (int off = 8; off; off >>= 1) ps += __shfl_xor(ps, off, 32);
      row_l[r] = row_l[r] * corr + ps;
      row_m[r] = nm;
#pragma unroll
      for (int nt = 0; nt < 4; nt++) o[nt][r] *= corr;
    }

    // C-layout P -> LDS -> A-layout fragment
#pragma unroll
    for (int nt = 0; nt < 4; nt++)
#pragma unroll
      for (int r = 0; r < 8; r++)
        Ps[wave][hi * 8 + r][nt * 16 + ln] = f2bf(sc[nt][r]);
    __syncthreads();

    // O += P * V  (A = P frag; B lane n = d column, K-dim = key)
#pragma unroll
    for (int ks = 0; ks < 2; ks++) {
      v16bf pa;
#pragma unroll
      for (int p = 0; p < 8; p++) {
        int kk = ks * 32 + ((p < 4) ? (hi * 8 + 2 * p) : (16 + hi * 8 + 2 * (p - 4)));
        pa[2 * p]     = Ps[wave][ln][kk];
        pa[2 * p + 1] = Ps[wave][ln][kk + 1];
      }
#pragma unroll
      for (int nt = 0; nt < 4; nt++) {
        v16bf bfr;
        const int d   = nt * 16 + ln;
        const int kst = ks * 32 + hi * 16;
#pragma unroll
        for (int j = 0; j < 16; j++) bfr[j] = Vt[d][(kst & ~31) + ((kst & 16) >> 1) + j + (j & 8)];
        o[nt] = __builtin_amdgcn_wmma_f32_16x16x32_bf16(
            false, pa, false, bfr, (short)0, o[nt], false, false);
      }
    }
    __syncthreads();
  }

  // Normalize and write ctx (bf16 flat [B*S, H])
#pragma unroll
  for (int nt = 0; nt < 4; nt++) {
    const int d = nt * 16 + ln;
#pragma unroll
    for (int r = 0; r < 8; r++) {
      const int m = q0 + wave * 16 + hi * 8 + r;
      const float v = o[nt][r] / row_l[r];
      ctx[(size_t)(b * S_ + m) * H_ + hcol + d] = f2bf(v);
    }
  }
}

// ---------------------------------------------------------------------------
// Host launch
// ---------------------------------------------------------------------------
extern "C" void kernel_launch(void* const* d_in, const int* in_sizes, int n_in,
                              void* d_out, int out_size, void* d_ws, size_t ws_size,
                              hipStream_t stream) {
  const float* x  = (const float*)d_in[0];
  const float* Wq = (const float*)d_in[1];
  const float* bq = (const float*)d_in[2];
  const float* Wk = (const float*)d_in[3];
  const float* bk = (const float*)d_in[4];
  const float* Wv = (const float*)d_in[5];
  const float* bv = (const float*)d_in[6];
  const float* Wo = (const float*)d_in[7];
  const float* bo = (const float*)d_in[8];
  float* out = (float*)d_out;

  // Workspace layout (bf16 scratch), ~92 MB total
  char* ws = (char*)d_ws;
  const size_t XB = (size_t)M_ * H_ * sizeof(bf16);   // 16 MB
  const size_t WB = (size_t)H_ * H_ * sizeof(bf16);   // 2 MB
  bf16* xb   = (bf16*)(ws);
  bf16* wqb  = (bf16*)(ws + XB);
  bf16* wkb  = (bf16*)(ws + XB + 1 * WB);
  bf16* wvb  = (bf16*)(ws + XB + 2 * WB);
  bf16* wob  = (bf16*)(ws + XB + 3 * WB);
  bf16* qb   = (bf16*)(ws + XB + 4 * WB);
  bf16* kb   = (bf16*)(ws + XB + 4 * WB + 1 * XB);
  bf16* vb   = (bf16*)(ws + XB + 4 * WB + 2 * XB);
  bf16* ctxb = (bf16*)(ws + XB + 4 * WB + 3 * XB);

  // 1) Convert inputs/weights to bf16
  cvt_f32_bf16<<<2048, 256, 0, stream>>>(x,  xb,  M_ * H_);
  cvt_f32_bf16<<<1024, 256, 0, stream>>>(Wq, wqb, H_ * H_);
  cvt_f32_bf16<<<1024, 256, 0, stream>>>(Wk, wkb, H_ * H_);
  cvt_f32_bf16<<<1024, 256, 0, stream>>>(Wv, wvb, H_ * H_);
  cvt_f32_bf16<<<1024, 256, 0, stream>>>(Wo, wob, H_ * H_);

  // 2) Q/K/V projections (bf16 out, flat [M,H])
  dim3 ggrid(M_ / 256, H_ / 64);
  gemm_bf16<0><<<ggrid, 256, 0, stream>>>(xb, wqb, bq, qb, M_, H_, H_);
  gemm_bf16<0><<<ggrid, 256, 0, stream>>>(xb, wkb, bk, kb, M_, H_, H_);
  gemm_bf16<0><<<ggrid, 256, 0, stream>>>(xb, wvb, bv, vb, M_, H_, H_);

  // 3) Attention (64 (b,h) pairs x 32 Q-tiles)
  dim3 agrid(B_ * NH_, S_ / 64);
  flash_attn<<<agrid, 128, 0, stream>>>(qb, kb, vb, ctxb);

  // 4) Output projection (f32 out -> d_out)
  gemm_bf16<1><<<ggrid, 256, 0, stream>>>(ctxb, wob, bo, out, M_, H_, H_);
}
